// MultiModalBlockchainModel_61804579389670
// MI455X (gfx1250) — matrix operand order repaired and checked by reading, not compiled, not run on hardware
//
#include <hip/hip_runtime.h>
#include <cmath>
#include <cstdint>
#include <cstddef>

// ---------------------------------------------------------------------------
// Model constants (from the reference)
// ---------------------------------------------------------------------------
constexpr int NN  = 100000;   // nodes
constexpr int NE  = 400000;   // edges
constexpr int HID = 128;      // hidden dim
constexpr float NEG_SLOPE = 0.2f;

typedef __bf16 bf16x16 __attribute__((ext_vector_type(16)));
typedef float  v8f     __attribute__((ext_vector_type(8)));
typedef int    v4i     __attribute__((ext_vector_type(4)));

#define GLOBAL_AS __attribute__((address_space(1)))
#define LDS_AS    __attribute__((address_space(3)))

__device__ __forceinline__ unsigned short f32_to_bf16(float f) {
  unsigned int u = __float_as_uint(f);
  unsigned int r = 0x7FFFu + ((u >> 16) & 1u);  // round-to-nearest-even
  return (unsigned short)((u + r) >> 16);
}

union FragBF { bf16x16 v; unsigned short u[16]; uint4 q[2]; };

// ---------------------------------------------------------------------------
// WMMA GEMM: C[M,N] = act(A[M,K] * B[K,N] + bias)
//   A fp32 row-major (converted to bf16 in-register)
//   B pre-converted bf16, TRANSPOSED: Bt[N][K]; staged per-block into LDS via
//   gfx1250 async global->LDS copies (ASYNCcnt path), padded stride K+8.
// Block = 256 threads = 8 waves; block tile 128(M) x 64(N); K step 32.
// Requires N%64==0, K%32==0, M%16==0.
// ---------------------------------------------------------------------------
__global__ __launch_bounds__(256) void gemm_bf16_wmma(
    const float* __restrict__ A, const unsigned short* __restrict__ Bt,
    const float* __restrict__ bias, float* __restrict__ C,
    int M, int N, int K, int relu)
{
  __shared__ unsigned short sB[64 * 136];      // 64 rows x (128+8) halves max

  const int lane = threadIdx.x & 31;
  const int wave = threadIdx.x >> 5;
  const int m0 = blockIdx.y * 128 + wave * 16;
  const int n0 = blockIdx.x * 64;
  const int KP = K + 8;                        // padded LDS row stride (halves)

  // ---- stage B panel (64 x K bf16) into LDS, 16B chunks ----
  {
    const int cpr = K >> 3;                    // 8-half chunks per row
    const int total = 64 * cpr;                // <= 1024
    for (int i = threadIdx.x; i < total; i += 256) {
      const int n = i / cpr;
      const int j = i - n * cpr;
      const unsigned short* g = Bt + (size_t)(n0 + n) * K + j * 8;
      unsigned short* l = sB + n * KP + j * 8;
#if __has_builtin(__builtin_amdgcn_global_load_async_to_lds_b128)
      __builtin_amdgcn_global_load_async_to_lds_b128(
          (GLOBAL_AS v4i*)g, (LDS_AS v4i*)l, 0, 0);
#else
      *(uint4*)l = *(const uint4*)g;
#endif
    }
#if __has_builtin(__builtin_amdgcn_global_load_async_to_lds_b128)
#if __has_builtin(__builtin_amdgcn_s_wait_asynccnt)
    __builtin_amdgcn_s_wait_asynccnt(0);
#else
    asm volatile("s_wait_asynccnt 0x0" ::: "memory");
#endif
#endif
    __syncthreads();
  }

  // clamp so every wave computes (keeps the barrier above uniform); stores are
  // guarded wave-uniformly below.  M%16==0 so mrow tiles never cross M.
  const int mrow = (m0 < M) ? m0 : (M - 16);

  const int r15    = lane & 15;
  const int half   = lane >> 4;                // 0 or 1
  const int ka_off = half * 8;                 // A: hi lanes hold K+8
  const int kb_off = half * 8;                 // B: hi lanes hold K+8

  v8f acc[4] = {};

  const float* arow = A + (size_t)(mrow + r15) * K;
  for (int kk = 0; kk < K; kk += 32) {
    // ---- A fragment: K chunks {0..7, 16..23} (+ka_off), 2 bf16 per VGPR ----
    FragBF a;
    {
      const float4 f0 = *(const float4*)(arow + kk + ka_off + 0);
      const float4 f1 = *(const float4*)(arow + kk + ka_off + 4);
      const float4 f2 = *(const float4*)(arow + kk + ka_off + 16);
      const float4 f3 = *(const float4*)(arow + kk + ka_off + 20);
      a.u[0]  = f32_to_bf16(f0.x); a.u[1]  = f32_to_bf16(f0.y);
      a.u[2]  = f32_to_bf16(f0.z); a.u[3]  = f32_to_bf16(f0.w);
      a.u[4]  = f32_to_bf16(f1.x); a.u[5]  = f32_to_bf16(f1.y);
      a.u[6]  = f32_to_bf16(f1.z); a.u[7]  = f32_to_bf16(f1.w);
      a.u[8]  = f32_to_bf16(f2.x); a.u[9]  = f32_to_bf16(f2.y);
      a.u[10] = f32_to_bf16(f2.z); a.u[11] = f32_to_bf16(f2.w);
      a.u[12] = f32_to_bf16(f3.x); a.u[13] = f32_to_bf16(f3.y);
      a.u[14] = f32_to_bf16(f3.z); a.u[15] = f32_to_bf16(f3.w);
    }
    // ---- 4 N-tiles of 16: B frag = K chunks {0..7, 16..23} (+kb_off) of column n (LDS) ----
#pragma unroll
    for (int t = 0; t < 4; ++t) {
      FragBF b;
      const unsigned short* lp = sB + (t * 16 + r15) * KP + kk + kb_off;
      b.q[0] = *(const uint4*)(lp);
      b.q[1] = *(const uint4*)(lp + 16);
      acc[t] = __builtin_amdgcn_wmma_f32_16x16x32_bf16(
          false, a.v, false, b.v, (short)0, acc[t], false, false);
    }
  }

  if (m0 < M) {                                // wave-uniform store guard
#pragma unroll
    for (int t = 0; t < 4; ++t) {
      const int col = n0 + t * 16 + r15;
      const float bc = bias ? bias[col] : 0.f;
#pragma unroll
      for (int i = 0; i < 8; ++i) {
        const int row = m0 + i + half * 8;
        float v = acc[t][i] + bc;
        if (relu) v = fmaxf(v, 0.f);
        C[(size_t)row * N + col] = v;
      }
    }
  }
}

// ---------------------------------------------------------------------------
// Weight fp32 [K][N] -> bf16 transposed [N][K]
// ---------------------------------------------------------------------------
__global__ void convert_transpose(const float* __restrict__ W,
                                  unsigned short* __restrict__ Bt, int K, int N) {
  int idx = blockIdx.x * blockDim.x + threadIdx.x;
  if (idx >= K * N) return;
  int n = idx / K, k = idx - n * K;
  Bt[idx] = f32_to_bf16(W[(size_t)k * N + n]);
}

// v[h][k] = sum_c W[k, h*128+c] * a[h][c]   (fold attention vec through W)
__global__ void fold_attn(const float* __restrict__ W, const float* __restrict__ a,
                          float* __restrict__ v, int H) {
  int idx = blockIdx.x * blockDim.x + threadIdx.x;    // h*128 + k
  if (idx >= H * HID) return;
  int h = idx >> 7, k = idx & 127;
  const float* wrow = W + (size_t)k * (H * HID) + h * HID;
  const float* ar = a + h * HID;
  float s = 0.f;
  for (int c = 0; c < HID; ++c) s += wrow[c] * ar[c];
  v[idx] = s;
}

// One wave per (n,h): al_s[n,h] = X[n]·vs[h], al_d[n,h] = X[n]·vd[h]
__global__ __launch_bounds__(256) void attn_logits_node(
    const float* __restrict__ X, const float* __restrict__ vs,
    const float* __restrict__ vd, float* __restrict__ als,
    float* __restrict__ ald, int rows, int H) {
  int gw = (int)((blockIdx.x * (size_t)blockDim.x + threadIdx.x) >> 5);
  int lane = threadIdx.x & 31;
  if (gw >= rows * H) return;
  int n = gw / H, h = gw - n * H;
  const float4 x = *(const float4*)(X + (size_t)n * HID + lane * 4);
  const float4 a = *(const float4*)(vs + (size_t)h * HID + lane * 4);
  const float4 b = *(const float4*)(vd + (size_t)h * HID + lane * 4);
  float ss = x.x * a.x + x.y * a.y + x.z * a.z + x.w * a.w;
  float sd = x.x * b.x + x.y * b.y + x.z * b.z + x.w * b.w;
#pragma unroll
  for (int off = 16; off > 0; off >>= 1) {
    ss += __shfl_xor(ss, off, 32);
    sd += __shfl_xor(sd, off, 32);
  }
  if (lane == 0) { als[gw] = ss; ald[gw] = sd; }
}

__global__ __launch_bounds__(256) void attn_logits_edge(
    const float* __restrict__ X, const float* __restrict__ ve,
    float* __restrict__ ale, int rows, int H) {
  int gw = (int)((blockIdx.x * (size_t)blockDim.x + threadIdx.x) >> 5);
  int lane = threadIdx.x & 31;
  if (gw >= rows * H) return;
  int n = gw / H, h = gw - n * H;
  const float4 x = *(const float4*)(X + (size_t)n * HID + lane * 4);
  const float4 a = *(const float4*)(ve + (size_t)h * HID + lane * 4);
  float s = x.x * a.x + x.y * a.y + x.z * a.z + x.w * a.w;
#pragma unroll
  for (int off = 16; off > 0; off >>= 1) s += __shfl_xor(s, off, 32);
  if (lane == 0) ale[gw] = s;
}

__global__ void fill_f32(float* __restrict__ p, float v, size_t n) {
  size_t i = (size_t)blockIdx.x * blockDim.x + threadIdx.x;
  if (i < n) p[i] = v;
}

// alpha = leakyrelu(al_s[src]+al_d[dst]+al_e); mx[dst] = max(...)
__global__ void edge_alpha_max(const int* __restrict__ src, const int* __restrict__ dst,
                               const float* __restrict__ als, const float* __restrict__ ald,
                               const float* __restrict__ ale, float* __restrict__ alpha,
                               float* __restrict__ mx, int E, int H) {
  int idx = blockIdx.x * blockDim.x + threadIdx.x;
  if (idx >= E * H) return;
  int e = idx / H, h = idx - e * H;
  float v = als[src[e] * H + h] + ald[dst[e] * H + h] + ale[idx];
  v = v > 0.f ? v : NEG_SLOPE * v;
  alpha[idx] = v;
  atomicMax(&mx[dst[e] * H + h], v);     // global_atomic_max_num_f32
}

__global__ void mx_finalize(float* __restrict__ mx, int n) {
  int i = blockIdx.x * blockDim.x + threadIdx.x;
  if (i >= n) return;
  float m = mx[i];
  if (m == -INFINITY) mx[i] = 0.f;       // mirrors jnp.where(isfinite(mx), mx, 0)
}

__global__ void edge_exp_sum(const int* __restrict__ dst, const float* __restrict__ mx,
                             float* __restrict__ alpha, float* __restrict__ sm,
                             int E, int H) {
  int idx = blockIdx.x * blockDim.x + threadIdx.x;
  if (idx >= E * H) return;
  int e = idx / H, h = idx - e * H;
  float ex = expf(alpha[idx] - mx[dst[e] * H + h]);
  alpha[idx] = ex;                        // reuse alpha buffer as ex
  atomicAdd(&sm[dst[e] * H + h], ex);
}

// w[e,h] = ex[e,h] / (sm[dst,h] + 1e-16)  (in place on the ex buffer)
__global__ void edge_weight(const int* __restrict__ dst, const float* __restrict__ sm,
                            float* __restrict__ w, int E, int H) {
  int idx = blockIdx.x * blockDim.x + threadIdx.x;
  if (idx >= E * H) return;
  int e = idx / H, h = idx - e * H;
  w[idx] = w[idx] / (sm[dst[e] * H + h] + 1e-16f);
}

// out[dst,h,c] += xs[src,h,c] * w[e,h]; one thread per (e,h,c/4)
__global__ __launch_bounds__(256) void scatter_messages(
    const int* __restrict__ src, const int* __restrict__ dst,
    const float* __restrict__ xs, const float* __restrict__ w,
    float* __restrict__ out, int E, int H) {
  size_t idx = (size_t)blockIdx.x * blockDim.x + threadIdx.x;
  size_t total = (size_t)E * H * (HID / 4);
  if (idx >= total) return;
  int c4 = (int)(idx & 31);               // HID/4 == 32
  size_t eh = idx >> 5;                   // e*H + h
  int e = (int)(eh / H);
  int h = (int)(eh - (size_t)e * H);
  int d = dst[e], s = src[e];
  float ww = w[eh];
  const float4 xv = *(const float4*)(xs + ((size_t)s * H + h) * HID + c4 * 4);
  float* o = out + ((size_t)d * H + h) * HID + c4 * 4;
  atomicAdd(o + 0, xv.x * ww);
  atomicAdd(o + 1, xv.y * ww);
  atomicAdd(o + 2, xv.z * ww);
  atomicAdd(o + 3, xv.w * ww);
}

__global__ void head_mean_bias_relu(const float* __restrict__ acc,
                                    const float* __restrict__ bias,
                                    float* __restrict__ dst, int rows, int H) {
  int idx = blockIdx.x * blockDim.x + threadIdx.x;
  if (idx >= rows * HID) return;
  int n = idx >> 7, c = idx & 127;
  float s = 0.f;
  for (int h = 0; h < H; ++h) s += acc[((size_t)n * H + h) * HID + c];
  dst[idx] = fmaxf(s * (1.f / H) + bias[c], 0.f);
}

// One wave per node: logits[n,0..1] = X[n] @ W2[128,2] + b2
__global__ __launch_bounds__(256) void classifier_logits(
    const float* __restrict__ X, const float* __restrict__ W2,
    const float* __restrict__ b2, float* __restrict__ logits, int rows) {
  int gw = (int)((blockIdx.x * (size_t)blockDim.x + threadIdx.x) >> 5);
  int lane = threadIdx.x & 31;
  if (gw >= rows) return;
  const float4 x = *(const float4*)(X + (size_t)gw * HID + lane * 4);
  int k = lane * 4;
  float s0 = x.x * W2[(k + 0) * 2 + 0] + x.y * W2[(k + 1) * 2 + 0] +
             x.z * W2[(k + 2) * 2 + 0] + x.w * W2[(k + 3) * 2 + 0];
  float s1 = x.x * W2[(k + 0) * 2 + 1] + x.y * W2[(k + 1) * 2 + 1] +
             x.z * W2[(k + 2) * 2 + 1] + x.w * W2[(k + 3) * 2 + 1];
#pragma unroll
  for (int off = 16; off > 0; off >>= 1) {
    s0 += __shfl_xor(s0, off, 32);
    s1 += __shfl_xor(s1, off, 32);
  }
  if (lane == 0) {
    logits[(size_t)gw * 2 + 0] = s0 + b2[0];
    logits[(size_t)gw * 2 + 1] = s1 + b2[1];
  }
}

// ---------------------------------------------------------------------------
// Host orchestration
// ---------------------------------------------------------------------------
extern "C" void kernel_launch(void* const* d_in, const int* in_sizes, int n_in,
                              void* d_out, int out_size, void* d_ws, size_t ws_size,
                              hipStream_t stream) {
  (void)in_sizes; (void)n_in; (void)out_size; (void)ws_size;
  static const int HEADS[3] = {4, 4, 1};

  // Inputs: setup_inputs() dict order; params dict in insertion order.
  const float* x    = (const float*)d_in[0];
  const int*   ei   = (const int*)d_in[1];
  const float* ea   = (const float*)d_in[2];
  const float* neW1 = (const float*)d_in[3];
  const float* neb1 = (const float*)d_in[4];
  const float* neW2 = (const float*)d_in[5];
  const float* neb2 = (const float*)d_in[6];
  const float* eeW1 = (const float*)d_in[7];
  const float* eeb1 = (const float*)d_in[8];
  const float* eeW2 = (const float*)d_in[9];
  const float* eeb2 = (const float*)d_in[10];
  const float *gW[3], *gWe[3], *gas[3], *gad[3], *gae[3], *gb[3];
  for (int li = 0; li < 3; ++li) {
    int b = 11 + li * 6;
    gW[li]  = (const float*)d_in[b + 0];
    gWe[li] = (const float*)d_in[b + 1];
    gas[li] = (const float*)d_in[b + 2];
    gad[li] = (const float*)d_in[b + 3];
    gae[li] = (const float*)d_in[b + 4];
    gb[li]  = (const float*)d_in[b + 5];
  }
  const float* cW1 = (const float*)d_in[29];
  const float* cb1 = (const float*)d_in[30];
  const float* cW2 = (const float*)d_in[31];
  const float* cb2 = (const float*)d_in[32];

  const int* srcI = ei;
  const int* dstI = ei + NE;

  float* logits = (float*)d_out;
  float* emb    = (float*)d_out + (size_t)NN * 2;

  // bump allocator over workspace
  char* wp = (char*)d_ws;
  auto bump = [&](size_t bytes) -> void* {
    void* r = (void*)wp;
    wp += (bytes + 255) & ~(size_t)255;
    return r;
  };
  float* hbuf   = (float*)bump((size_t)NN * HID * 4);       // node embedding
  float* ebuf   = (float*)bump((size_t)NE * HID * 4);       // edge embedding
  float* xsbuf  = (float*)bump((size_t)NN * 4 * HID * 4);   // xs, max H=4
  float* bigbuf = (float*)bump((size_t)NE * HID * 4);       // h1/e1/c1/out (shared)
  float* als    = (float*)bump((size_t)NN * 4 * 4);
  float* ald    = (float*)bump((size_t)NN * 4 * 4);
  float* ale    = (float*)bump((size_t)NE * 4 * 4);
  float* alpha  = (float*)bump((size_t)NE * 4 * 4);
  float* mx     = (float*)bump((size_t)NN * 4 * 4);
  float* sm     = (float*)bump((size_t)NN * 4 * 4);
  unsigned short* neW1t = (unsigned short*)bump(128 * 128 * 2);
  unsigned short* neW2t = (unsigned short*)bump(128 * 128 * 2);
  unsigned short* eeW1t = (unsigned short*)bump(32 * 128 * 2);
  unsigned short* eeW2t = (unsigned short*)bump(128 * 128 * 2);
  unsigned short* gWt[3];
  gWt[0] = (unsigned short*)bump(128 * 512 * 2);
  gWt[1] = (unsigned short*)bump(128 * 512 * 2);
  gWt[2] = (unsigned short*)bump(128 * 128 * 2);
  unsigned short* cW1t = (unsigned short*)bump(128 * 128 * 2);
  float *vsv[3], *vdv[3], *vev[3];
  for (int li = 0; li < 3; ++li) {
    vsv[li] = (float*)bump(4 * 128 * 4);
    vdv[li] = (float*)bump(4 * 128 * 4);
    vev[li] = (float*)bump(4 * 128 * 4);
  }

  auto convT = [&](const float* W, unsigned short* Bt, int K, int N) {
    int tot = K * N;
    convert_transpose<<<(tot + 255) / 256, 256, 0, stream>>>(W, Bt, K, N);
  };
  convT(neW1, neW1t, 128, 128);
  convT(neW2, neW2t, 128, 128);
  convT(eeW1, eeW1t, 32, 128);
  convT(eeW2, eeW2t, 128, 128);
  convT(gW[0], gWt[0], 128, 512);
  convT(gW[1], gWt[1], 128, 512);
  convT(gW[2], gWt[2], 128, 128);
  convT(cW1, cW1t, 128, 128);

  for (int li = 0; li < 3; ++li) {
    int H = HEADS[li];
    int tot = H * HID;
    fold_attn<<<(tot + 127) / 128, 128, 0, stream>>>(gW[li], gas[li], vsv[li], H);
    fold_attn<<<(tot + 127) / 128, 128, 0, stream>>>(gW[li], gad[li], vdv[li], H);
    fold_attn<<<(tot + 127) / 128, 128, 0, stream>>>(gWe[li], gae[li], vev[li], H);
  }

  auto gemm = [&](const float* A, const unsigned short* Bt, const float* bias,
                  float* C, int M, int N, int K, int relu) {
    dim3 g((unsigned)(N / 64), (unsigned)((M + 127) / 128));
    gemm_bf16_wmma<<<g, 256, 0, stream>>>(A, Bt, bias, C, M, N, K, relu);
  };

  // node encoder: h = (relu(x @ W1 + b1)) @ W2 + b2
  gemm(x, neW1t, neb1, bigbuf, NN, 128, 128, 1);
  gemm(bigbuf, neW2t, neb2, hbuf, NN, 128, 128, 0);
  // edge encoder
  gemm(ea, eeW1t, eeb1, bigbuf, NE, 128, 32, 1);
  gemm(bigbuf, eeW2t, eeb2, ebuf, NE, 128, 128, 0);

  for (int li = 0; li < 3; ++li) {
    const int H = HEADS[li];
    const float* hin = hbuf;
    float* hout = (li == 2) ? emb : hbuf;

    // xs = h @ W  [NN, H*128]
    gemm(hin, gWt[li], nullptr, xsbuf, NN, H * 128, 128, 0);

    // attention logits via folded vectors
    {
      size_t nw = (size_t)NN * H;
      attn_logits_node<<<(unsigned)((nw * 32 + 255) / 256), 256, 0, stream>>>(
          hin, vsv[li], vdv[li], als, ald, NN, H);
      size_t ew = (size_t)NE * H;
      attn_logits_edge<<<(unsigned)((ew * 32 + 255) / 256), 256, 0, stream>>>(
          ebuf, vev[li], ale, NE, H);
    }

    // segment softmax init
    {
      size_t n = (size_t)NN * H;
      fill_f32<<<(unsigned)((n + 255) / 256), 256, 0, stream>>>(mx, -INFINITY, n);
      (void)hipMemsetAsync(sm, 0, n * 4, stream);
      (void)hipMemsetAsync(bigbuf, 0, (size_t)NN * H * HID * 4, stream);
    }

    int eh = NE * H;
    edge_alpha_max<<<(eh + 255) / 256, 256, 0, stream>>>(srcI, dstI, als, ald, ale,
                                                         alpha, mx, NE, H);
    mx_finalize<<<(NN * H + 255) / 256, 256, 0, stream>>>(mx, NN * H);
    edge_exp_sum<<<(eh + 255) / 256, 256, 0, stream>>>(dstI, mx, alpha, sm, NE, H);
    edge_weight<<<(eh + 255) / 256, 256, 0, stream>>>(dstI, sm, alpha, NE, H);

    size_t total = (size_t)NE * H * (HID / 4);
    scatter_messages<<<(unsigned)((total + 255) / 256), 256, 0, stream>>>(
        srcI, dstI, xsbuf, alpha, bigbuf, NE, H);

    head_mean_bias_relu<<<(NN * HID + 255) / 256, 256, 0, stream>>>(bigbuf, gb[li],
                                                                    hout, NN, H);
  }

  // classifier: logits = relu(emb @ cW1 + cb1) @ cW2 + cb2
  gemm(emb, cW1t, cb1, bigbuf, NN, 128, 128, 1);
  classifier_logits<<<(unsigned)(((size_t)NN * 32 + 255) / 256), 256, 0, stream>>>(
      bigbuf, cW2, cb2, logits, NN);
}
